// TissueTransMIL_77300821394015
// MI455X (gfx1250) — matrix operand order repaired and checked by reading, not compiled, not run on hardware
//
#include <hip/hip_runtime.h>

#define BATCH 2
#define HN    16384
#define NTOK  16385
#define NP    16640
#define PADL  255
#define MLM   256
#define LSEG  65
#define DIMC  512
#define INDIM 1536
#define NH    8
#define DH    64
#define BH    16          // BATCH*NH
#define SCALE_F 0.125f    // 64^-0.5

typedef __attribute__((ext_vector_type(16))) __bf16 v16bf;
typedef __attribute__((ext_vector_type(8)))  float  v8f;

struct alignas(16) U128 { unsigned int x[4]; };
union BF16Frag { v16bf v; U128 q[2]; };

// ---------------------------------------------------------------- reductions
__device__ inline float block_reduce256(float v, float* red, int t, bool domax) {
  red[t] = v; __syncthreads();
  for (int s = 128; s > 0; s >>= 1) {
    if (t < s) red[t] = domax ? fmaxf(red[t], red[t + s]) : (red[t] + red[t + s]);
    __syncthreads();
  }
  float r = red[0];
  __syncthreads();
  return r;
}

// ---------------------------------------------------------------- WMMA GEMM
// C[M x N] (f32) = A[M x K] (bf16 row-major) @ Bt[N x K]^T (bf16 row-major N x K)
// Register-blocked (16x64 strip per wave) + software-pipelined K loop (2 frag sets).
// Requires K % 64 == 0 (true for all call sites: 64/256/512/1536/16640).
// epilogue: 0 = none, 1 = +bias, 2 = +bias & relu.  Batched via strides.
__device__ inline void load_frags(BF16Frag& a, BF16Frag& b0, BF16Frag& b1,
                                  BF16Frag& b2, BF16Frag& b3,
                                  const __bf16* __restrict__ aptr,
                                  const __bf16* __restrict__ p0,
                                  const __bf16* __restrict__ p1,
                                  const __bf16* __restrict__ p2,
                                  const __bf16* __restrict__ p3, int kk) {
  a.q[0]  = *reinterpret_cast<const U128*>(aptr + kk);
  a.q[1]  = *reinterpret_cast<const U128*>(aptr + kk + 8);
  b0.q[0] = *reinterpret_cast<const U128*>(p0 + kk);
  b0.q[1] = *reinterpret_cast<const U128*>(p0 + kk + 8);
  b1.q[0] = *reinterpret_cast<const U128*>(p1 + kk);
  b1.q[1] = *reinterpret_cast<const U128*>(p1 + kk + 8);
  b2.q[0] = *reinterpret_cast<const U128*>(p2 + kk);
  b2.q[1] = *reinterpret_cast<const U128*>(p2 + kk + 8);
  b3.q[0] = *reinterpret_cast<const U128*>(p3 + kk);
  b3.q[1] = *reinterpret_cast<const U128*>(p3 + kk + 8);
}

__global__ __launch_bounds__(256) void wmma_gemm_kernel(
    const __bf16* __restrict__ A, const __bf16* __restrict__ Bt,
    float* __restrict__ C, const float* __restrict__ bias,
    int M, int N, int K, long sA, long sB, long sC, int epi)
{
  int wave = threadIdx.x >> 5;
  int lane = threadIdx.x & 31;
  int tile_m = blockIdx.y * 8 + wave;
  if (tile_m * 16 >= M) return;              // wave-uniform: EXEC stays all-ones
  int n0 = blockIdx.x * 64;                  // 64-wide N strip per wave
  long batch = blockIdx.z;
  const __bf16* Ab = A + batch * sA;
  const __bf16* Bb = Bt + batch * sB;
  float* Cb = C + batch * sC;

  int am = tile_m * 16 + (lane & 15);
  int ak = (lane >> 4) * 16;                 // A: lane halves K = 0..15 / 16..31
  int bk = (lane >> 4) * 16;                 // B: lane halves K = 0..15 / 16..31

  const __bf16* aptr  = Ab + (long)am * K + ak;
  const __bf16* bptr0 = Bb + (long)(n0 + 0  + (lane & 15)) * K + bk;
  const __bf16* bptr1 = Bb + (long)(n0 + 16 + (lane & 15)) * K + bk;
  const __bf16* bptr2 = Bb + (long)(n0 + 32 + (lane & 15)) * K + bk;
  const __bf16* bptr3 = Bb + (long)(n0 + 48 + (lane & 15)) * K + bk;

  v8f acc0 = {}, acc1 = {}, acc2 = {}, acc3 = {};
  BF16Frag a0, c00, c01, c02, c03;           // set 0
  BF16Frag a1, c10, c11, c12, c13;           // set 1
  load_frags(a0, c00, c01, c02, c03, aptr, bptr0, bptr1, bptr2, bptr3, 0);
  for (int kk = 0; kk < K; kk += 64) {
    if (kk + 512 < K) {
      __builtin_prefetch(aptr + kk + 512, 0, 3);
      __builtin_prefetch(bptr0 + kk + 512, 0, 3);
      __builtin_prefetch(bptr2 + kk + 512, 0, 3);
    }
    // stage the second half-step's fragments, then compute on set 0
    load_frags(a1, c10, c11, c12, c13, aptr, bptr0, bptr1, bptr2, bptr3, kk + 32);
    acc0 = __builtin_amdgcn_wmma_f32_16x16x32_bf16(false, a0.v, false, c00.v, (short)0, acc0, false, false);
    acc1 = __builtin_amdgcn_wmma_f32_16x16x32_bf16(false, a0.v, false, c01.v, (short)0, acc1, false, false);
    acc2 = __builtin_amdgcn_wmma_f32_16x16x32_bf16(false, a0.v, false, c02.v, (short)0, acc2, false, false);
    acc3 = __builtin_amdgcn_wmma_f32_16x16x32_bf16(false, a0.v, false, c03.v, (short)0, acc3, false, false);
    // stage the next step's set 0 while computing on set 1
    if (kk + 64 < K)
      load_frags(a0, c00, c01, c02, c03, aptr, bptr0, bptr1, bptr2, bptr3, kk + 64);
    acc0 = __builtin_amdgcn_wmma_f32_16x16x32_bf16(false, a1.v, false, c10.v, (short)0, acc0, false, false);
    acc1 = __builtin_amdgcn_wmma_f32_16x16x32_bf16(false, a1.v, false, c11.v, (short)0, acc1, false, false);
    acc2 = __builtin_amdgcn_wmma_f32_16x16x32_bf16(false, a1.v, false, c12.v, (short)0, acc2, false, false);
    acc3 = __builtin_amdgcn_wmma_f32_16x16x32_bf16(false, a1.v, false, c13.v, (short)0, acc3, false, false);
  }

  int row0 = tile_m * 16 + (lane >> 4) * 8;
  int cl = lane & 15;
  v8f accs[4] = {acc0, acc1, acc2, acc3};
#pragma unroll
  for (int j = 0; j < 4; ++j) {
    int col = n0 + j * 16 + cl;
    float bv = (epi >= 1 && bias) ? bias[col] : 0.f;
#pragma unroll
    for (int r = 0; r < 8; ++r) {
      float v = accs[j][r] + bv;
      if (epi == 2) v = fmaxf(v, 0.f);
      Cb[(long)(row0 + r) * N + col] = v;
    }
  }
}

// ---------------------------------------------------------------- f32 tiled GEMM (pinv chain)
// mode 0: C=A@B   mode 1: C=alpha*I - A@B   mode 2: C=0.25*(A@B)
__global__ __launch_bounds__(256) void sgemm_f32_kernel(
    const float* __restrict__ A, const float* __restrict__ B, float* __restrict__ C,
    int M, int N, int K, long sA, long sB, long sC, int mode, float alpha)
{
  __shared__ float As[16][17], Bs[16][17];
  int tx = threadIdx.x & 15, ty = threadIdx.x >> 4;
  int bm = blockIdx.y * 16, bn = blockIdx.x * 16;
  long bt = blockIdx.z;
  const float* Ab = A + bt * sA;
  const float* Bb = B + bt * sB;
  float* Cb = C + bt * sC;
  float acc = 0.f;
  for (int k0 = 0; k0 < K; k0 += 16) {
    As[ty][tx] = Ab[(long)(bm + ty) * K + k0 + tx];
    Bs[ty][tx] = Bb[(long)(k0 + ty) * N + bn + tx];
    __syncthreads();
#pragma unroll
    for (int kk = 0; kk < 16; ++kk) acc += As[ty][kk] * Bs[kk][tx];
    __syncthreads();
  }
  int r = bm + ty, c = bn + tx;
  float v = (mode == 1) ? (((r == c) ? alpha : 0.f) - acc)
          : (mode == 2) ? 0.25f * acc : acc;
  Cb[(long)r * N + c] = v;
}

// ---------------------------------------------------------------- elementwise helpers
__global__ void cast_bf16_kernel(const float* __restrict__ in, __bf16* __restrict__ out, long n) {
  long i = (long)blockIdx.x * 256 + threadIdx.x;
  if (i < n) out[i] = (__bf16)in[i];
}

// out[b][c][r] = (bf16) in[b][r][c]   (in: batch x R x Ccol)
__global__ void transpose_cast_kernel(const float* __restrict__ in, __bf16* __restrict__ out,
                                      int R, int Ccol, long total) {
  long i = (long)blockIdx.x * 256 + threadIdx.x;
  if (i >= total) return;
  int c = (int)(i % Ccol);
  long t = i / Ccol;
  int r = (int)(t % R);
  long b = t / R;
  out[b * (long)R * Ccol + (long)c * R + r] = (__bf16)in[i];
}

__global__ void fill_cls_kernel(const float* __restrict__ cls, float* __restrict__ h) {
  int i = blockIdx.x * 256 + threadIdx.x;
  if (i >= BATCH * DIMC) return;
  int b = i >> 9, c = i & 511;
  h[(long)b * NTOK * DIMC + c] = cls[c];
}

__global__ void assemble_h_kernel(const float* __restrict__ tmp, float* __restrict__ h) {
  long i = (long)blockIdx.x * 256 + threadIdx.x;
  if (i >= (long)BATCH * HN * DIMC) return;
  int c = (int)(i & 511);
  long t = i >> 9;
  long row = t % HN, b = t / HN;
  h[((long)b * NTOK + 1 + row) * DIMC + c] = tmp[i];
}

__global__ void copy_row0_kernel(const float* __restrict__ src, float* __restrict__ dst) {
  int i = blockIdx.x * 256 + threadIdx.x;
  if (i >= BATCH * DIMC) return;
  int b = i >> 9, c = i & 511;
  dst[(long)b * NTOK * DIMC + c] = src[(long)b * NTOK * DIMC + c];
}

__global__ void add_out_kernel(float* __restrict__ h, const float* __restrict__ op) {
  long i = (long)blockIdx.x * 256 + threadIdx.x;
  if (i >= (long)BATCH * NTOK * DIMC) return;
  int c = (int)(i & 511);
  long t = i >> 9;
  long row = t % NTOK, b = t / NTOK;
  h[i] += op[((long)b * NP + PADL + row) * DIMC + c];
}

// ---------------------------------------------------------------- layernorm -> padded bf16
__global__ __launch_bounds__(256) void layernorm_pad_kernel(
    const float* __restrict__ h, const float* __restrict__ g, const float* __restrict__ bb,
    __bf16* __restrict__ out)
{
  int row = blockIdx.x;                    // b*NP + n
  int b = row / NP, n = row % NP;
  __bf16* orow = out + (long)row * DIMC;
  int t = threadIdx.x;
  if (n < PADL) {
    for (int c = t; c < DIMC; c += 256) orow[c] = (__bf16)0.f;
    return;
  }
  const float* x = h + ((long)b * NTOK + (n - PADL)) * DIMC;
  __shared__ float red[256];
  float s = 0.f;
  for (int c = t; c < DIMC; c += 256) s += x[c];
  float mu = block_reduce256(s, red, t, false) * (1.f / DIMC);
  float s2 = 0.f;
  for (int c = t; c < DIMC; c += 256) { float d = x[c] - mu; s2 += d * d; }
  float rstd = rsqrtf(block_reduce256(s2, red, t, false) * (1.f / DIMC) + 1e-5f);
  for (int c = t; c < DIMC; c += 256)
    orow[c] = (__bf16)((x[c] - mu) * rstd * g[c] + bb[c]);
}

// ---------------------------------------------------------------- landmarks
__global__ void landmark_kernel(const float* __restrict__ qkv,
                                float* __restrict__ ql, float* __restrict__ kl) {
  long i = (long)blockIdx.x * 256 + threadIdx.x;
  if (i >= (long)BH * MLM * DH) return;
  int d = (int)(i & 63);
  long t = i >> 6;
  int j = (int)(t & 255);
  int bh = (int)(t >> 8);
  int b = bh >> 3, hh = bh & 7;
  const float* qb = qkv + ((long)b * NP + (long)j * LSEG) * (3 * DIMC) + hh * DH + d;
  float sq = 0.f, sk = 0.f;
  for (int s = 0; s < LSEG; ++s) { sq += qb[(long)s * 3 * DIMC]; sk += qb[(long)s * 3 * DIMC + DIMC]; }
  ql[i] = sq * (SCALE_F / LSEG);
  kl[i] = sk * (1.f / LSEG);
}

// ---------------------------------------------------------------- fused softmax kernels
__global__ __launch_bounds__(256) void softmax256_kernel(float* __restrict__ x) {
  long row = blockIdx.x;
  float* xr = x + row * MLM;
  int t = threadIdx.x;
  __shared__ float red[256];
  float v = xr[t];
  float mx = block_reduce256(v, red, t, true);
  float e = __expf(v - mx);
  float sm = block_reduce256(e, red, t, false);
  xr[t] = e / sm;
}

// attn1 = softmax_m( (q*scale) @ k_l^T ) -> bf16 [bh][NP][256]
__global__ __launch_bounds__(256) void attn1_kernel(const float* __restrict__ qkv,
                                                    const float* __restrict__ kl,
                                                    __bf16* __restrict__ attn1) {
  long row = blockIdx.x;                  // bh*NP + n
  int bh = (int)(row / NP);
  int n = (int)(row % NP);
  int b = bh >> 3, hh = bh & 7;
  __shared__ float qrow[DH];
  __shared__ float red[256];
  int t = threadIdx.x;
  if (t < DH) qrow[t] = qkv[((long)b * NP + n) * (3 * DIMC) + hh * DH + t] * SCALE_F;
  __syncthreads();
  const float* klh = kl + (long)bh * MLM * DH + (long)t * DH;
  float dot = 0.f;
#pragma unroll 8
  for (int d = 0; d < DH; ++d) dot += qrow[d] * klh[d];
  float mx = block_reduce256(dot, red, t, true);
  float e = __expf(dot - mx);
  float sm = block_reduce256(e, red, t, false);
  attn1[row * MLM + t] = (__bf16)(e / sm);
}

// attn3 = softmax_N( q_l @ k^T ) -> bf16 [bh][256][NP]
__global__ __launch_bounds__(256) void attn3_kernel(const float* __restrict__ qkv,
                                                    const float* __restrict__ ql,
                                                    __bf16* __restrict__ attn3) {
  int row = blockIdx.x;                  // bh*256 + jm
  int bh = row >> 8, jm = row & 255;
  int b = bh >> 3, hh = bh & 7;
  __shared__ float qrow[DH];
  __shared__ float red[256];
  int t = threadIdx.x;
  if (t < DH) qrow[t] = ql[((long)bh * MLM + jm) * DH + t];
  __syncthreads();
  float mx = -3.4e38f;
  for (int n = t; n < NP; n += 256) {
    const float* krow = qkv + ((long)b * NP + n) * (3 * DIMC) + DIMC + hh * DH;
    float dot = 0.f;
#pragma unroll 8
    for (int d = 0; d < DH; ++d) dot += qrow[d] * krow[d];
    mx = fmaxf(mx, dot);
  }
  mx = block_reduce256(mx, red, t, true);
  float sm = 0.f;
  for (int n = t; n < NP; n += 256) {
    const float* krow = qkv + ((long)b * NP + n) * (3 * DIMC) + DIMC + hh * DH;
    float dot = 0.f;
#pragma unroll 8
    for (int d = 0; d < DH; ++d) dot += qrow[d] * krow[d];
    sm += __expf(dot - mx);
  }
  sm = block_reduce256(sm, red, t, false);
  float inv = 1.f / sm;
  __bf16* orow = attn3 + (long)bh * MLM * NP + (long)jm * NP;
  for (int n = t; n < NP; n += 256) {
    const float* krow = qkv + ((long)b * NP + n) * (3 * DIMC) + DIMC + hh * DH;
    float dot = 0.f;
#pragma unroll 8
    for (int d = 0; d < DH; ++d) dot += qrow[d] * krow[d];
    orow[n] = (__bf16)(__expf(dot - mx) * inv);
  }
}

// ---------------------------------------------------------------- vT (per-head v transpose, bf16)
__global__ __launch_bounds__(256) void vT_kernel(const float* __restrict__ qkv,
                                                 __bf16* __restrict__ vT) {
  __shared__ __bf16 tile[64][65];
  int bh = blockIdx.y;
  int b = bh >> 3, hh = bh & 7;
  int n0 = blockIdx.x * 64;
#pragma unroll
  for (int p = 0; p < 16; ++p) {
    int e = p * 256 + threadIdx.x;
    int d = e & 63, ni = e >> 6;
    tile[ni][d] = (__bf16)qkv[((long)b * NP + n0 + ni) * (3 * DIMC) + 2 * DIMC + hh * DH + d];
  }
  __syncthreads();
#pragma unroll
  for (int p = 0; p < 16; ++p) {
    int e = p * 256 + threadIdx.x;
    int nn = e & 63, d = e >> 6;
    vT[((long)bh * DH + d) * NP + n0 + nn] = tile[nn][d];
  }
}

// ---------------------------------------------------------------- pinv helpers
__global__ void zero_scal_kernel(float* scal) { if (threadIdx.x < 2) scal[threadIdx.x] = 0.f; }

__global__ __launch_bounds__(256) void pinv_norm_kernel(const float* __restrict__ x, float* scal) {
  int bh = blockIdx.x, t = threadIdx.x;
  const float* xb = x + (long)bh * MLM * MLM;
  float rs = 0.f, cs = 0.f;
  for (int j = 0; j < MLM; ++j) {
    rs += fabsf(xb[(long)t * MLM + j]);
    cs += fabsf(xb[(long)j * MLM + t]);
  }
  __shared__ float red[256];
  float mr = block_reduce256(rs, red, t, true);
  if (t == 0) atomicMax(reinterpret_cast<int*>(&scal[0]), __float_as_int(mr));
  float mc = block_reduce256(cs, red, t, true);
  if (t == 0) atomicMax(reinterpret_cast<int*>(&scal[1]), __float_as_int(mc));
}

__global__ void pinv_init_kernel(const float* __restrict__ x, const float* __restrict__ scal,
                                 float* __restrict__ z) {
  long i = (long)blockIdx.x * 256 + threadIdx.x;
  if (i >= (long)BH * MLM * MLM) return;
  int j = (int)(i & 255);
  int r = (int)((i >> 8) & 255);
  long bh = i >> 16;
  z[i] = x[(bh << 16) + ((long)j << 8) + r] / (scal[0] * scal[1]);
}

__global__ void ident_sub_kernel(float* __restrict__ out, const float* __restrict__ in,
                                 float alpha, long total) {
  long i = (long)blockIdx.x * 256 + threadIdx.x;
  if (i >= total) return;
  int c = (int)(i & 255);
  int r = (int)((i >> 8) & 255);
  out[i] = ((r == c) ? alpha : 0.f) - in[i];
}

// ---------------------------------------------------------------- combine: attn_out + depthwise res conv
__global__ void combine_kernel(const float* __restrict__ aout, const float* __restrict__ qkv,
                               const float* __restrict__ resk, __bf16* __restrict__ comb) {
  long i = (long)blockIdx.x * 256 + threadIdx.x;
  if (i >= (long)BATCH * NP * DIMC) return;
  int c = (int)(i & 511);
  long t = i >> 9;
  int n = (int)(t % NP);
  int b = (int)(t / NP);
  int hh = c >> 6, d = c & 63;
  float r = 0.f;
#pragma unroll 4
  for (int tt = 0; tt < 33; ++tt) {
    int nn = n + tt - 16;
    if (nn >= 0 && nn < NP)
      r += resk[hh * 33 + tt] * qkv[((long)b * NP + nn) * (3 * DIMC) + 2 * DIMC + hh * DH + d];
  }
  float a = aout[(((long)(b * NH + hh)) * NP + n) * DH + d];
  comb[i] = (__bf16)(a + r);
}

// ---------------------------------------------------------------- PPEG (depthwise 7/5/3 convs)
__global__ void ppeg_kernel(const float* __restrict__ hin, float* __restrict__ hout,
                            const float* __restrict__ k7, const float* __restrict__ b7,
                            const float* __restrict__ k5, const float* __restrict__ b5,
                            const float* __restrict__ k3, const float* __restrict__ b3) {
  long i = (long)blockIdx.x * 256 + threadIdx.x;
  if (i >= (long)BATCH * DIMC * HN) return;
  int ch = (int)(i & 511);
  long t = i >> 9;
  int x = (int)(t & 127); t >>= 7;
  int y = (int)(t & 127); t >>= 7;
  int b = (int)t;
  const float* base = hin + (long)b * NTOK * DIMC + DIMC;  // skip cls row
  float acc = base[(long)(y * 128 + x) * DIMC + ch] + b7[ch] + b5[ch] + b3[ch];
  for (int dy = -3; dy <= 3; ++dy) {
    int yy = y + dy;
    if (yy < 0 || yy > 127) continue;
    for (int dx = -3; dx <= 3; ++dx) {
      int xx = x + dx;
      if (xx < 0 || xx > 127) continue;
      float v = base[(long)(yy * 128 + xx) * DIMC + ch];
      acc += v * k7[(ch * 7 + (dy + 3)) * 7 + (dx + 3)];
      if (dy >= -2 && dy <= 2 && dx >= -2 && dx <= 2)
        acc += v * k5[(ch * 5 + (dy + 2)) * 5 + (dx + 2)];
      if (dy >= -1 && dy <= 1 && dx >= -1 && dx <= 1)
        acc += v * k3[(ch * 3 + (dy + 1)) * 3 + (dx + 1)];
    }
  }
  hout[(long)b * NTOK * DIMC + DIMC + (long)(y * 128 + x) * DIMC + ch] = acc;
}

// ---------------------------------------------------------------- final LN(row0) @ W2 + b2
__global__ __launch_bounds__(256) void final_kernel(const float* __restrict__ h,
                                                    const float* __restrict__ g,
                                                    const float* __restrict__ bb,
                                                    const float* __restrict__ W2,
                                                    const float* __restrict__ b2,
                                                    float* __restrict__ out) {
  int b = blockIdx.x, t = threadIdx.x;
  const float* x = h + (long)b * NTOK * DIMC;  // row 0 (cls token)
  __shared__ float red[256];
  float s = 0.f;
  for (int c = t; c < DIMC; c += 256) s += x[c];
  float mu = block_reduce256(s, red, t, false) * (1.f / DIMC);
  float s2 = 0.f;
  for (int c = t; c < DIMC; c += 256) { float d = x[c] - mu; s2 += d * d; }
  float rstd = rsqrtf(block_reduce256(s2, red, t, false) * (1.f / DIMC) + 1e-5f);
  float dp = 0.f;
  for (int c = t; c < DIMC; c += 256)
    dp += ((x[c] - mu) * rstd * g[c] + bb[c]) * W2[c];
  dp = block_reduce256(dp, red, t, false);
  if (t == 0) out[b] = dp + b2[0];
}

// ================================================================ host side
static inline unsigned gdiv(long a, long b) { return (unsigned)((a + b - 1) / b); }

static void launch_wmma_gemm(const __bf16* A, const __bf16* Bt, float* C, const float* bias,
                             int M, int N, int K, long sA, long sB, long sC, int batch,
                             int epi, hipStream_t st) {
  dim3 g(N / 64, (M / 16 + 7) / 8, batch);   // each wave: 16x64 strip
  wmma_gemm_kernel<<<g, 256, 0, st>>>(A, Bt, C, bias, M, N, K, sA, sB, sC, epi);
}

static void launch_sgemm(const float* A, const float* B, float* C, int M, int N, int K,
                         long sA, long sB, long sC, int batch, int mode, float alpha,
                         hipStream_t st) {
  dim3 g(N / 16, M / 16, batch);
  sgemm_f32_kernel<<<g, 256, 0, st>>>(A, B, C, M, N, K, sA, sB, sC, mode, alpha);
}

struct Scratch {
  __bf16 *wbuf, *xln, *ql_bf, *kl_bf, *attn1, *attn3, *vT, *pt3vT, *comb;
  float *qkv, *ql, *kl, *attn2, *zA, *zB, *xz, *wt, *tt, *t3v, *pt3v, *aout, *outproj, *scal;
};

static void run_nystrom(float* h, const float* lng, const float* lnb,
                        const float* Wqkv, const float* Wout, const float* bout,
                        const float* resk, const Scratch& S, hipStream_t st) {
  // x = LN(h), zero-padded (255 leading rows), bf16
  layernorm_pad_kernel<<<BATCH * NP, 256, 0, st>>>(h, lng, lnb, S.xln);
  // qkv = x @ Wqkv
  transpose_cast_kernel<<<gdiv((long)DIMC * 3 * DIMC, 256), 256, 0, st>>>(
      Wqkv, S.wbuf, DIMC, 3 * DIMC, (long)DIMC * 3 * DIMC);
  launch_wmma_gemm(S.xln, S.wbuf, S.qkv, nullptr, BATCH * NP, 3 * DIMC, DIMC, 0, 0, 0, 1, 0, st);
  // landmarks (q scaled inside)
  landmark_kernel<<<gdiv((long)BH * MLM * DH, 256), 256, 0, st>>>(S.qkv, S.ql, S.kl);
  cast_bf16_kernel<<<gdiv((long)BH * MLM * DH, 256), 256, 0, st>>>(S.ql, S.ql_bf, (long)BH * MLM * DH);
  cast_bf16_kernel<<<gdiv((long)BH * MLM * DH, 256), 256, 0, st>>>(S.kl, S.kl_bf, (long)BH * MLM * DH);
  // attn2 = softmax(q_l @ k_l^T)
  launch_wmma_gemm(S.ql_bf, S.kl_bf, S.attn2, nullptr, MLM, MLM, DH,
                   (long)MLM * DH, (long)MLM * DH, (long)MLM * MLM, BH, 0, st);
  softmax256_kernel<<<BH * MLM, 256, 0, st>>>(S.attn2);
  // fused softmax attns + v transpose
  attn1_kernel<<<BH * NP, 256, 0, st>>>(S.qkv, S.kl, S.attn1);
  attn3_kernel<<<BH * MLM, 256, 0, st>>>(S.qkv, S.ql, S.attn3);
  vT_kernel<<<dim3(NP / 64, BH), 256, 0, st>>>(S.qkv, S.vT);
  // t3v = attn3 @ v
  launch_wmma_gemm(S.attn3, S.vT, S.t3v, nullptr, MLM, DH, NP,
                   (long)MLM * NP, (long)DH * NP, (long)MLM * DH, BH, 0, st);
  // Moore-Penrose pinv of attn2 (6 Newton-Schulz iterations, f32)
  zero_scal_kernel<<<1, 32, 0, st>>>(S.scal);
  pinv_norm_kernel<<<BH, 256, 0, st>>>(S.attn2, S.scal);
  pinv_init_kernel<<<gdiv((long)BH * MLM * MLM, 256), 256, 0, st>>>(S.attn2, S.scal, S.zA);
  long sq = (long)MLM * MLM;
  float* zc = S.zA; float* zn = S.zB;
  for (int it = 0; it < 6; ++it) {
    launch_sgemm(S.attn2, zc, S.xz, MLM, MLM, MLM, sq, sq, sq, BH, 0, 0.f, st);
    ident_sub_kernel<<<gdiv((long)BH * sq, 256), 256, 0, st>>>(S.wt, S.xz, 7.f, (long)BH * sq);
    launch_sgemm(S.xz, S.wt, S.tt, MLM, MLM, MLM, sq, sq, sq, BH, 0, 0.f, st);
    ident_sub_kernel<<<gdiv((long)BH * sq, 256), 256, 0, st>>>(S.wt, S.tt, 15.f, (long)BH * sq);
    launch_sgemm(S.xz, S.wt, S.tt, MLM, MLM, MLM, sq, sq, sq, BH, 0, 0.f, st);
    ident_sub_kernel<<<gdiv((long)BH * sq, 256), 256, 0, st>>>(S.wt, S.tt, 13.f, (long)BH * sq);
    launch_sgemm(zc, S.wt, zn, MLM, MLM, MLM, sq, sq, sq, BH, 2, 0.f, st);
    float* sw = zc; zc = zn; zn = sw;
  }
  // pt3v = pinv @ t3v ; transpose to bf16 for WMMA B operand
  launch_sgemm(zc, S.t3v, S.pt3v, MLM, DH, MLM, sq, (long)MLM * DH, (long)MLM * DH, BH, 0, 0.f, st);
  transpose_cast_kernel<<<gdiv((long)BH * MLM * DH, 256), 256, 0, st>>>(
      S.pt3v, S.pt3vT, MLM, DH, (long)BH * MLM * DH);
  // attn_out = attn1 @ pt3v
  launch_wmma_gemm(S.attn1, S.pt3vT, S.aout, nullptr, NP, DH, MLM,
                   (long)NP * MLM, (long)DH * MLM, (long)NP * DH, BH, 0, st);
  // + depthwise 33-tap residual on v, merge heads, cast bf16
  combine_kernel<<<gdiv((long)BATCH * NP * DIMC, 256), 256, 0, st>>>(S.aout, S.qkv, resk, S.comb);
  // out = comb @ Wout + bout ; h += out[:, -NTOK:]
  transpose_cast_kernel<<<gdiv((long)DIMC * DIMC, 256), 256, 0, st>>>(
      Wout, S.wbuf, DIMC, DIMC, (long)DIMC * DIMC);
  launch_wmma_gemm(S.comb, S.wbuf, S.outproj, bout, BATCH * NP, DIMC, DIMC, 0, 0, 0, 1, 1, st);
  add_out_kernel<<<gdiv((long)BATCH * NTOK * DIMC, 256), 256, 0, st>>>(h, S.outproj);
}

extern "C" void kernel_launch(void* const* d_in, const int* in_sizes, int n_in,
                              void* d_out, int out_size, void* d_ws, size_t ws_size,
                              hipStream_t stream) {
  (void)in_sizes; (void)n_in; (void)out_size; (void)ws_size;
  const float* features = (const float*)d_in[0];
  const float* W1    = (const float*)d_in[1];
  const float* b1    = (const float*)d_in[2];
  const float* cls   = (const float*)d_in[3];
  const float* ln1_g = (const float*)d_in[4];
  const float* ln1_b = (const float*)d_in[5];
  const float* Wqkv1 = (const float*)d_in[6];
  const float* Wout1 = (const float*)d_in[7];
  const float* bout1 = (const float*)d_in[8];
  const float* resk1 = (const float*)d_in[9];
  const float* ln2_g = (const float*)d_in[10];
  const float* ln2_b = (const float*)d_in[11];
  const float* Wqkv2 = (const float*)d_in[12];
  const float* Wout2 = (const float*)d_in[13];
  const float* bout2 = (const float*)d_in[14];
  const float* resk2 = (const float*)d_in[15];
  const float* k7    = (const float*)d_in[16];
  const float* b7c   = (const float*)d_in[17];
  const float* k5    = (const float*)d_in[18];
  const float* b5c   = (const float*)d_in[19];
  const float* k3    = (const float*)d_in[20];
  const float* b3c   = (const float*)d_in[21];
  const float* lnf_g = (const float*)d_in[22];
  const float* lnf_b = (const float*)d_in[23];
  const float* W2    = (const float*)d_in[24];
  const float* b2    = (const float*)d_in[25];

  unsigned char* base = (unsigned char*)d_ws;
  size_t off = 0;
  auto alloc = [&](size_t bytes) -> void* {
    void* p = base + off;
    off = (off + bytes + 255) & ~(size_t)255;
    return p;
  };

  __bf16* feat_bf = (__bf16*)alloc((size_t)BATCH * HN * INDIM * 2);
  float*  tmp_h   = (float*)alloc((size_t)BATCH * HN * DIMC * 4);
  float*  hA      = (float*)alloc((size_t)BATCH * NTOK * DIMC * 4);
  float*  hB      = (float*)alloc((size_t)BATCH * NTOK * DIMC * 4);

  Scratch S;
  S.wbuf   = (__bf16*)alloc((size_t)INDIM * DIMC * 2);
  S.xln    = (__bf16*)alloc((size_t)BATCH * NP * DIMC * 2);
  S.qkv    = (float*)alloc((size_t)BATCH * NP * 3 * DIMC * 4);
  S.vT     = (__bf16*)alloc((size_t)BH * DH * NP * 2);
  S.ql     = (float*)alloc((size_t)BH * MLM * DH * 4);
  S.kl     = (float*)alloc((size_t)BH * MLM * DH * 4);
  S.ql_bf  = (__bf16*)alloc((size_t)BH * MLM * DH * 2);
  S.kl_bf  = (__bf16*)alloc((size_t)BH * MLM * DH * 2);
  S.attn1  = (__bf16*)alloc((size_t)BH * NP * MLM * 2);
  S.attn3  = (__bf16*)alloc((size_t)BH * MLM * NP * 2);
  S.attn2  = (float*)alloc((size_t)BH * MLM * MLM * 4);
  S.zA     = (float*)alloc((size_t)BH * MLM * MLM * 4);
  S.zB     = (float*)alloc((size_t)BH * MLM * MLM * 4);
  S.xz     = (float*)alloc((size_t)BH * MLM * MLM * 4);
  S.wt     = (float*)alloc((size_t)BH * MLM * MLM * 4);
  S.tt     = (float*)alloc((size_t)BH * MLM * MLM * 4);
  S.t3v    = (float*)alloc((size_t)BH * MLM * DH * 4);
  S.pt3v   = (float*)alloc((size_t)BH * MLM * DH * 4);
  S.pt3vT  = (__bf16*)alloc((size_t)BH * DH * MLM * 2);
  S.aout   = (float*)alloc((size_t)BH * NP * DH * 4);
  S.comb   = (__bf16*)alloc((size_t)BATCH * NP * DIMC * 2);
  S.outproj= (float*)alloc((size_t)BATCH * NP * DIMC * 4);
  S.scal   = (float*)alloc(256);

  // h = relu(features @ W1 + b1)
  cast_bf16_kernel<<<gdiv((long)BATCH * HN * INDIM, 256), 256, 0, stream>>>(
      features, feat_bf, (long)BATCH * HN * INDIM);
  transpose_cast_kernel<<<gdiv((long)INDIM * DIMC, 256), 256, 0, stream>>>(
      W1, S.wbuf, INDIM, DIMC, (long)INDIM * DIMC);
  launch_wmma_gemm(feat_bf, S.wbuf, tmp_h, b1, BATCH * HN, DIMC, INDIM, 0, 0, 0, 1, 2, stream);
  // prepend cls token
  fill_cls_kernel<<<gdiv(BATCH * DIMC, 256), 256, 0, stream>>>(cls, hA);
  assemble_h_kernel<<<gdiv((long)BATCH * HN * DIMC, 256), 256, 0, stream>>>(tmp_h, hA);
  // block 1
  run_nystrom(hA, ln1_g, ln1_b, Wqkv1, Wout1, bout1, resk1, S, stream);
  // PPEG
  copy_row0_kernel<<<gdiv(BATCH * DIMC, 256), 256, 0, stream>>>(hA, hB);
  ppeg_kernel<<<gdiv((long)BATCH * DIMC * HN, 256), 256, 0, stream>>>(
      hA, hB, k7, b7c, k5, b5c, k3, b3c);
  // block 2
  run_nystrom(hB, ln2_g, ln2_b, Wqkv2, Wout2, bout2, resk2, S, stream);
  // final head
  final_kernel<<<BATCH, 256, 0, stream>>>(hB, lnf_g, lnf_b, W2, b2, (float*)d_out);
}
